// BidLSTM_3977139716149
// MI455X (gfx1250) — matrix operand order repaired and checked by reading, not compiled, not run on hardware
//
#include <hip/hip_runtime.h>
#include <stdint.h>

// Problem sizes (match reference)
constexpr int kB = 16, kT = 256, kV = 32000, kE = 256, kH = 512;
constexpr int kBT = kB * kT;        // 4096
constexpr int kH2 = 2 * kH;         // 1024
constexpr int kG4H = 4 * kH;        // 2048

typedef __attribute__((ext_vector_type(16))) __bf16 v16bf;
typedef __attribute__((ext_vector_type(8)))  float  v8f;
typedef __attribute__((ext_vector_type(4)))  unsigned uint4v;
typedef __attribute__((ext_vector_type(4)))  unsigned tdm_g0_t;
typedef __attribute__((ext_vector_type(8)))  int      tdm_g1_t;
typedef __attribute__((ext_vector_type(4)))  int      tdm_g23_t;

// ---------------------------------------------------------------------------
// WMMA helper (CDNA5 wave32, 16x16x32 bf16 -> f32)
// ---------------------------------------------------------------------------
__device__ __forceinline__ v8f wmma_bf16(v16bf a, v16bf b, v8f c) {
  return __builtin_amdgcn_wmma_f32_16x16x32_bf16(
      /*neg_a=*/false, a, /*neg_b=*/false, b,
      /*c_mod=*/(short)0, c, /*reuse_a=*/false, /*reuse_b=*/false);
}

// A fragment, ISA 16-bit A 16x32 layout: lane holds row (lane&15);
// lanes 0-15: K {k0+0..7, k0+16..23}; lanes 16-31: +8.
__device__ __forceinline__ v16bf load_fragA(const __bf16* __restrict__ base,
                                            int row, int ld, int k0, int lane) {
  const int hi = (lane & 16) ? 8 : 0;
  const __bf16* p = base + (size_t)row * ld + k0 + hi;
  union { v16bf v; uint4v q[2]; } f;
  f.q[0] = *(const uint4v*)(p);        // K = k0+hi   .. k0+hi+7
  f.q[1] = *(const uint4v*)(p + 16);   // K = k0+hi+16.. k0+hi+23
  return f.v;
}

// B fragment, ISA 16-bit B 32x16 layout: lane holds column (lane&15);
// lanes 0-15: K {k0+0..7, k0+16..23}; lanes 16-31: +8.  N-major source ->
// two 16B vector loads per lane.
__device__ __forceinline__ v16bf load_fragB(const __bf16* __restrict__ baseT,
                                            int n, int ld, int k0, int lane) {
  const int hi = (lane & 16) ? 8 : 0;
  const __bf16* p = baseT + (size_t)n * ld + k0 + hi;
  union { v16bf v; uint4v q[2]; } f;
  f.q[0] = *(const uint4v*)(p);
  f.q[1] = *(const uint4v*)(p + 16);
  return f.v;
}

// ---------------------------------------------------------------------------
// Kernel 1: tiled fp32 -> bf16 transpose: src (K x N) -> dst (N x K) bf16.
// ---------------------------------------------------------------------------
__global__ __launch_bounds__(256) void transpose_f32_bf16_kernel(
    const float* __restrict__ src, __bf16* __restrict__ dst, int K, int N) {
  __shared__ float tile[32][33];
  const int kb = blockIdx.y * 32, nb = blockIdx.x * 32;
  const int tx = threadIdx.x & 31, ty = threadIdx.x >> 5;   // ty 0..7
#pragma unroll
  for (int i = ty; i < 32; i += 8)
    tile[i][tx] = src[(size_t)(kb + i) * N + nb + tx];
  __syncthreads();
#pragma unroll
  for (int i = ty; i < 32; i += 8)
    dst[(size_t)(nb + i) * K + kb + tx] = (__bf16)tile[tx][i];
}

// ---------------------------------------------------------------------------
// Kernel 2: embedding gather -> bf16 x[b*T+t][e]
// ---------------------------------------------------------------------------
__global__ void embed_kernel(const int* __restrict__ idx,
                             const float* __restrict__ emb,
                             __bf16* __restrict__ xbf) {
  int row = blockIdx.x;                  // b*T + t
  int e = threadIdx.x;                   // 0..255
  int token = idx[row];
  xbf[(size_t)row * kE + e] = (__bf16)emb[(size_t)token * kE + e];
}

// ---------------------------------------------------------------------------
// Kernel 3: xg = x @ Wx + b  for both directions.
// ---------------------------------------------------------------------------
__global__ __launch_bounds__(256) void xg_gemm_kernel(
    const __bf16* __restrict__ xbf,    // (4096 x 256) row = b*T+t
    const __bf16* __restrict__ wxT,    // 2 x (2048 x 256)
    const float* __restrict__ bias_f,  // (4,H) flat 2048
    const float* __restrict__ bias_b,
    float* __restrict__ xg)            // 2 x 4096 x 2048
{
  const int dir = blockIdx.z;
  const int m0 = blockIdx.x * 128;
  const int n0 = blockIdx.y * 128;
  const int lane = threadIdx.x & 31, wave = threadIdx.x >> 5;
  const int wm = wave >> 2, wn = wave & 3;

  const __bf16* WT = wxT + (size_t)dir * kG4H * kE;
  const float* bias = dir ? bias_b : bias_f;
  float* out = xg + (size_t)dir * kBT * kG4H;

  int srow[4];
#pragma unroll
  for (int mi = 0; mi < 4; ++mi) {
    int gr = m0 + wm * 64 + mi * 16 + (lane & 15);
    int t = gr >> 4, b = gr & 15;
    int tt = dir ? (kT - 1 - t) : t;
    srow[mi] = b * kT + tt;
  }

  v8f acc[4][2];
#pragma unroll
  for (int mi = 0; mi < 4; ++mi)
#pragma unroll
    for (int ni = 0; ni < 2; ++ni) acc[mi][ni] = (v8f)0.f;

  for (int kt = 0; kt < kE / 32; ++kt) {
    int k0 = kt * 32;
    v16bf a[4], b[2];
#pragma unroll
    for (int mi = 0; mi < 4; ++mi)
      a[mi] = load_fragA(xbf, srow[mi], kE, k0, lane);
#pragma unroll
    for (int ni = 0; ni < 2; ++ni) {
      int n = n0 + wn * 32 + ni * 16 + (lane & 15);
      b[ni] = load_fragB(WT, n, kE, k0, lane);
    }
#pragma unroll
    for (int mi = 0; mi < 4; ++mi)
#pragma unroll
      for (int ni = 0; ni < 2; ++ni)
        acc[mi][ni] = wmma_bf16(a[mi], b[ni], acc[mi][ni]);
  }

  const int hi8 = (lane >> 4) & 1;
#pragma unroll
  for (int mi = 0; mi < 4; ++mi) {
#pragma unroll
    for (int ni = 0; ni < 2; ++ni) {
      int col = n0 + wn * 32 + ni * 16 + (lane & 15);
      float bv = bias[col];
#pragma unroll
      for (int r = 0; r < 8; ++r) {
        int row = m0 + wm * 64 + mi * 16 + r + hi8 * 8;
        out[(size_t)row * kG4H + col] = acc[mi][ni][r] + bv;
      }
    }
  }
}

// ---------------------------------------------------------------------------
// Kernel 4: recurrent scan.
// ---------------------------------------------------------------------------
__global__ __launch_bounds__(128) void lstm_scan_kernel(
    const __bf16* __restrict__ whT,   // 2 x (2048 x 512)
    const float* __restrict__ xg,     // 2 x 4096 x 2048
    __bf16* __restrict__ hbuf,        // 2(pp) x 2(dir) x 16 x 512
    __bf16* __restrict__ hcat,        // (B*T) x 1024
    unsigned* __restrict__ syncc)     // 2 x T
{
  const int dir = blockIdx.x >> 4;
  const int slice = blockIdx.x & 15;
  const int hcol0 = slice * 32;
  const int tid = threadIdx.x, lane = tid & 31, wave = tid >> 5;  // gate = wave

  __shared__ float gates[16 * 128];   // [b][gate*32 + j]
  __shared__ float cst[16 * 32];      // cell state for owned columns
  for (int i = tid; i < 512; i += 128) cst[i] = 0.f;
  __syncthreads();

  const __bf16* WT = whT + (size_t)dir * kG4H * kH;
  const float* xgd = xg + (size_t)dir * kBT * kG4H;
  const int mrow = lane & 15;
  const int hi8 = (lane >> 4) & 1;

  for (int t = 0; t < kT; ++t) {
    const __bf16* hprev = hbuf + ((size_t)(t & 1) * 2 + dir) * kB * kH;
    __bf16* hnext = hbuf + ((size_t)((t + 1) & 1) * 2 + dir) * kB * kH;

    v8f acc[2] = {(v8f)0.f, (v8f)0.f};
    for (int kt = 0; kt < kH / 32; ++kt) {
      int k0 = kt * 32;
      v16bf a = load_fragA(hprev, mrow, kH, k0, lane);
#pragma unroll
      for (int ni = 0; ni < 2; ++ni) {
        int n = wave * kH + hcol0 + ni * 16 + (lane & 15);
        v16bf bfr = load_fragB(WT, n, kH, k0, lane);
        acc[ni] = wmma_bf16(a, bfr, acc[ni]);
      }
    }
#pragma unroll
    for (int ni = 0; ni < 2; ++ni) {
      int ncol = hcol0 + ni * 16 + (lane & 15);
#pragma unroll
      for (int r = 0; r < 8; ++r) {
        int b = r + hi8 * 8;
        float g = acc[ni][r] +
                  xgd[((size_t)t * 16 + b) * kG4H + wave * kH + ncol];
        gates[b * 128 + wave * 32 + (ncol - hcol0)] = g;
      }
    }
    __syncthreads();

    for (int i = tid; i < 512; i += 128) {
      int b = i >> 5, j = i & 31;
      float gi = gates[b * 128 + 0 * 32 + j];
      float gf = gates[b * 128 + 1 * 32 + j];
      float go = gates[b * 128 + 2 * 32 + j];
      float gc = gates[b * 128 + 3 * 32 + j];
      float iv = 1.f / (1.f + __expf(-gi));
      float fv = 1.f / (1.f + __expf(-gf));
      float ov = 1.f / (1.f + __expf(-go));
      float cv = fv * cst[i] + iv * tanhf(gc);
      cst[i] = cv;
      float hv = ov * tanhf(cv);
      __bf16 hb = (__bf16)hv;
      hnext[b * kH + hcol0 + j] = hb;
      int tout = dir ? (kT - 1 - t) : t;
      hcat[((size_t)b * kT + tout) * kH2 + dir * kH + hcol0 + j] = hb;
    }

    __threadfence();
    __syncthreads();
    if (tid == 0) {
      __hip_atomic_fetch_add(&syncc[dir * kT + t], 1u, __ATOMIC_ACQ_REL,
                             __HIP_MEMORY_SCOPE_AGENT);
      while (__hip_atomic_load(&syncc[dir * kT + t], __ATOMIC_ACQUIRE,
                               __HIP_MEMORY_SCOPE_AGENT) < 16u)
        __builtin_amdgcn_s_sleep(1);
    }
    __syncthreads();
  }
}

// ---------------------------------------------------------------------------
// Kernel 5: FC GEMM  out = hcat(4096x1024) @ fc_w^T(32000x1024) + fc_b.
// A panel staged with per-lane global_load_async_to_lds_b128 (ASYNCcnt);
// B panel staged with a single Tensor Data Mover descriptor per K-step
// (TENSORcnt), issued by wave 0.  Double-buffered LDS, 8 waves, wave tile
// 64x32 (8 WMMA accumulators).
// ---------------------------------------------------------------------------
#define FC_BM 128
#define FC_BN 128
#define FC_BK 32

__device__ __forceinline__ void async_b128(unsigned lds_addr, const void* g) {
  asm volatile("global_load_async_to_lds_b128 %0, %1, off"
               :: "v"(lds_addr), "v"((unsigned long long)(uintptr_t)g)
               : "memory");
}

// TDM 2D tile load: tile_k x tile_n elements (2B each), tile rows contiguous
// (length tile_k) with row stride stride_k elements; packed into LDS.
// D# fields per CDNA5 ISA 8.3/8.4 (count=1, type=2 "image", data_size=2B).
__device__ __forceinline__ void tdm_load_2d(unsigned lds_addr, const void* gaddr,
                                            unsigned tile_k, unsigned tile_n,
                                            unsigned tensor_k, unsigned tensor_n,
                                            unsigned stride_k) {
  unsigned long long ga = (unsigned long long)(uintptr_t)gaddr;
  tdm_g0_t g0;
  g0.x = 1u;                                           // count=1 (valid D#)
  g0.y = lds_addr;                                     // lds_addr [63:32]
  g0.z = (unsigned)(ga & 0xFFFFFFFFu);                 // global_addr[31:0]
  g0.w = (unsigned)((ga >> 32) & 0x01FFFFFFu)          // global_addr[56:32]
         | 0x80000000u;                                // type=2 at [127:126]
  tdm_g1_t g1;
  g1[0] = (int)(1u << 16);                             // data_size=1 -> 2 bytes
  g1[1] = (int)((tensor_k & 0xFFFFu) << 16);           // tensor_dim0[15:0]
  g1[2] = (int)(((tensor_k >> 16) & 0xFFFFu) |
                ((tensor_n & 0xFFFFu) << 16));         // dim0 hi | dim1 lo
  g1[3] = (int)(((tensor_n >> 16) & 0xFFFFu) |
                ((tile_k & 0xFFFFu) << 16));           // dim1 hi | tile_dim0
  g1[4] = (int)(tile_n & 0xFFFFu);                     // tile_dim1 (dim2=0)
  g1[5] = (int)stride_k;                               // tensor_dim0_stride lo
  g1[6] = 0;                                           // stride hi
  g1[7] = 0;
  tdm_g23_t z4 = {0, 0, 0, 0};                         // 2D: groups 2/3 unused
  tdm_g1_t  z8 = {0, 0, 0, 0, 0, 0, 0, 0};
  __builtin_amdgcn_tensor_load_to_lds(g0, g1, z4, z4, z8, 0);
}

__global__ __launch_bounds__(256) void fc_gemm_kernel(
    const __bf16* __restrict__ Abf,   // hcat 4096 x 1024
    const __bf16* __restrict__ BT,    // fc_w^T 32000 x 1024 (bf16)
    const float* __restrict__ bias,   // 32000
    float* __restrict__ out)          // 4096 x 32000
{
  __shared__ __bf16 lA[2][FC_BM * FC_BK];   // [m][k]
  __shared__ __bf16 lB[2][FC_BN * FC_BK];   // [n][k]

  const int K = kH2, N = kV;
  const int m0 = blockIdx.x * FC_BM;
  const int n0 = blockIdx.y * FC_BN;
  const int tid = threadIdx.x;
  const int lane = tid & 31, wave = tid >> 5;
  const int wm = wave >> 2, wn = wave & 3;    // 2 x 4 wave grid

  v8f acc[4][2];
#pragma unroll
  for (int mi = 0; mi < 4; ++mi)
#pragma unroll
    for (int ni = 0; ni < 2; ++ni) acc[mi][ni] = (v8f)0.f;

  // A tile: 128 rows x 32 cols bf16 = 512 x 16B chunks, 2 per thread (async).
  // B tile: one TDM descriptor (wave 0), 32(k) x 128(n) elements.
  auto issue = [&](int buf, int kt) {
    int k0 = kt * FC_BK;
#pragma unroll
    for (int i = 0; i < 2; ++i) {
      int id = tid * 2 + i;
      int r = id >> 2, c = id & 3;
      unsigned ldsa = (unsigned)(uintptr_t)&lA[buf][r * FC_BK + c * 8];
      async_b128(ldsa, Abf + (size_t)(m0 + r) * K + k0 + c * 8);
    }
    if (wave == 0) {
      unsigned ldsb = (unsigned)(uintptr_t)&lB[buf][0];
      tdm_load_2d(ldsb, BT + (size_t)n0 * K + k0,
                  /*tile_k=*/FC_BK, /*tile_n=*/FC_BN,
                  /*tensor_k=*/K, /*tensor_n=*/kV, /*stride_k=*/K);
    }
  };

  issue(0, 0);
  const int NK = K / FC_BK;   // 32
  for (int kt = 0; kt < NK; ++kt) {
    asm volatile("s_wait_asynccnt 0x0" ::: "memory");  // own A stages landed
    if (wave == 0) __builtin_amdgcn_s_wait_tensorcnt(0);  // B panel landed
    __syncthreads();                                   // everyone's landed
    if (kt + 1 < NK) issue((kt + 1) & 1, kt + 1);      // prefetch next panel

    const __bf16* As = lA[kt & 1];
    const __bf16* Bs = lB[kt & 1];
    v16bf a[4], b[2];
#pragma unroll
    for (int mi = 0; mi < 4; ++mi)
      a[mi] = load_fragA(As, wm * 64 + mi * 16 + (lane & 15), FC_BK, 0, lane);
#pragma unroll
    for (int ni = 0; ni < 2; ++ni)
      b[ni] = load_fragB(Bs, wn * 32 + ni * 16 + (lane & 15), FC_BK, 0, lane);
#pragma unroll
    for (int mi = 0; mi < 4; ++mi)
#pragma unroll
      for (int ni = 0; ni < 2; ++ni)
        acc[mi][ni] = wmma_bf16(a[mi], b[ni], acc[mi][ni]);
  }

  const int hi8 = (lane >> 4) & 1;
#pragma unroll
  for (int mi = 0; mi < 4; ++mi) {
#pragma unroll
    for (int ni = 0; ni < 2; ++ni) {
      int col = n0 + wn * 32 + ni * 16 + (lane & 15);
      float bv = bias[col];
#pragma unroll
      for (int r = 0; r < 8; ++r) {
        int row = m0 + wm * 64 + mi * 16 + r + hi8 * 8;
        out[(size_t)row * N + col] = acc[mi][ni][r] + bv;
      }
    }
  }
}

// ---------------------------------------------------------------------------
// Host side
// ---------------------------------------------------------------------------
extern "C" void kernel_launch(void* const* d_in, const int* in_sizes, int n_in,
                              void* d_out, int out_size, void* d_ws,
                              size_t ws_size, hipStream_t stream) {
  (void)in_sizes; (void)n_in; (void)out_size; (void)ws_size;

  const int*   inputs = (const int*)  d_in[0];   // (B,T) int32
  const float* emb    = (const float*)d_in[1];   // (V,E)
  const float* Wh_f   = (const float*)d_in[2];   // (4,H,H)
  const float* Wx_f   = (const float*)d_in[3];   // (4,E,H)
  const float* b_f    = (const float*)d_in[4];   // (4,H)
  const float* Wh_b   = (const float*)d_in[5];
  const float* Wx_b   = (const float*)d_in[6];
  const float* b_b    = (const float*)d_in[7];
  const float* fc_w   = (const float*)d_in[8];   // (2H,V)
  const float* fc_b   = (const float*)d_in[9];   // (V)
  float* out = (float*)d_out;                    // (B,T,V)

  size_t off = 0;
  auto carve = [&](size_t bytes) {
    void* p = (char*)d_ws + off;
    off += (bytes + 255) & ~(size_t)255;
    return p;
  };
  __bf16* xbf  = (__bf16*)carve((size_t)kBT * kE * 2);              // 2 MB
  __bf16* wxT  = (__bf16*)carve((size_t)2 * kG4H * kE * 2);         // 2 MB
  __bf16* whT  = (__bf16*)carve((size_t)2 * kG4H * kH * 2);         // 4 MB
  __bf16* fcwT = (__bf16*)carve((size_t)kV * kH2 * 2);              // 65.5 MB
  float*  xg   = (float*) carve((size_t)2 * kBT * kG4H * 4);        // 67 MB
  __bf16* hcat = (__bf16*)carve((size_t)kBT * kH2 * 2);             // 8 MB
  __bf16* hbuf = (__bf16*)carve((size_t)2 * 2 * kB * kH * 2);       // 64 KB
  unsigned* syncc = (unsigned*)carve((size_t)2 * kT * 4);           // 2 KB

  (void)hipMemsetAsync(hbuf, 0, (size_t)2 * 2 * kB * kH * 2, stream);
  (void)hipMemsetAsync(syncc, 0, (size_t)2 * kT * 4, stream);

  // Transpose all GEMM B-operands to N-major bf16 (per gate / per direction).
  for (int g = 0; g < 4; ++g) {
    transpose_f32_bf16_kernel<<<dim3(kH / 32, kE / 32), 256, 0, stream>>>(
        Wx_f + (size_t)g * kE * kH, wxT + (size_t)g * kH * kE, kE, kH);
    transpose_f32_bf16_kernel<<<dim3(kH / 32, kE / 32), 256, 0, stream>>>(
        Wx_b + (size_t)g * kE * kH, wxT + (size_t)(4 + g) * kH * kE, kE, kH);
    transpose_f32_bf16_kernel<<<dim3(kH / 32, kH / 32), 256, 0, stream>>>(
        Wh_f + (size_t)g * kH * kH, whT + (size_t)g * kH * kH, kH, kH);
    transpose_f32_bf16_kernel<<<dim3(kH / 32, kH / 32), 256, 0, stream>>>(
        Wh_b + (size_t)g * kH * kH, whT + (size_t)(4 + g) * kH * kH, kH, kH);
  }
  transpose_f32_bf16_kernel<<<dim3(kV / 32, kH2 / 32), 256, 0, stream>>>(
      fc_w, fcwT, kH2, kV);

  embed_kernel<<<kBT, kE, 0, stream>>>(inputs, emb, xbf);

  xg_gemm_kernel<<<dim3(kBT / 128, kG4H / 128, 2), 256, 0, stream>>>(
      xbf, wxT, b_f, b_b, xg);

  lstm_scan_kernel<<<32, 128, 0, stream>>>(whT, xg, hbuf, hcat, syncc);

  fc_gemm_kernel<<<dim3(kBT / FC_BM, kV / FC_BN), 256, 0, stream>>>(
      hcat, fcwT, fc_b, out);
}
